// TinyModelQuantized_48395691491530
// MI455X (gfx1250) — compile-verified, matrix operand order checked by reading, not run on hardware
//
#include <hip/hip_runtime.h>
#include <hip/hip_bf16.h>
#include <stdint.h>

typedef __attribute__((ext_vector_type(8))) int v8i;
typedef __attribute__((ext_vector_type(4))) int v4i;

#define BATCH    4096
#define IN_DIM   8192
#define HID_DIM  8192
#define OUT_DIM  8192
#define W_SCALE   0.1f
#define OUT_SCALE 0.1f

#define BK 128   // bytes of K consumed per mainloop iteration (2 WMMA k-steps)

// gfx1250 async global->LDS copy path (ASYNCcnt), if the toolchain exposes it.
#if defined(__has_builtin)
# if __has_builtin(__builtin_amdgcn_global_load_async_to_lds_b128) && \
     __has_builtin(__builtin_amdgcn_s_wait_asynccnt)
#  define USE_ASYNC_COPY 1
# endif
#endif
#ifndef USE_ASYNC_COPY
# define USE_ASYNC_COPY 0
#endif

#if USE_ASYNC_COPY
typedef __attribute__((address_space(1))) v4i v4i_as1;   // global
typedef __attribute__((address_space(3))) v4i v4i_as3;   // LDS
#endif

__device__ __forceinline__ float requant(float v) {
    return fminf(fmaxf(rintf(v), -128.0f), 127.0f);
}

// C[M,N] = requant( (A[M,K] int8) x (W[N,K] int8)^T * scale )
// 128x128 tile per 256-thread block; 8 waves, each wave owns 16 M-rows x 128 N-cols.
__global__ __launch_bounds__(256) void gemm_i8_wmma(
    const int8_t* __restrict__ A,      // [M,K] row-major
    const int8_t* __restrict__ W,      // [N,K] row-major
    int8_t* __restrict__ C,            // [M,N]
    int M, int N, int K,
    const float* __restrict__ scale_ptr, float scale_mul)
{
#if USE_ASYNC_COPY
    __shared__ __align__(16) int8_t lA[2][128 * BK];
    __shared__ __align__(16) int8_t lB[2][128 * BK];
#else
    __shared__ __align__(16) int8_t lA1[128 * BK];
    __shared__ __align__(16) int8_t lB1[128 * BK];
#endif

    const int tid  = threadIdx.x;
    const int wave = tid >> 5;
    const int lane = tid & 31;
    const int bm = blockIdx.y * 128;
    const int bn = blockIdx.x * 128;

    const float scale = (scale_ptr ? scale_ptr[0] : 1.0f) * scale_mul;

    v8i acc[8];
    const v8i vzero = {0, 0, 0, 0, 0, 0, 0, 0};
#pragma unroll
    for (int i = 0; i < 8; ++i) acc[i] = vzero;

    // cooperative-copy addressing: 256 threads x 64B each covers one 128x128B tile
    const int cprow = tid >> 1;          // 0..127
    const int cpcol = (tid & 1) * 64;    // 0 or 64
    const int8_t* gA = A + (size_t)(bm + cprow) * K + cpcol;
    const int8_t* gB = W + (size_t)(bn + cprow) * K + cpcol;
    const int ldsoff = cprow * BK + cpcol;

    const int khalf = (lane >> 4) * 8;
    const int frow  = lane & 15;

    // load one 16x64 8-bit fragment (A or B) from an LDS tile at byte `base`
    auto load_frag = [&](const int8_t* t, int base) {
        v8i f;
        int2 q0 = *(const int2*)&t[base +  0];
        int2 q1 = *(const int2*)&t[base + 16];
        int2 q2 = *(const int2*)&t[base + 32];
        int2 q3 = *(const int2*)&t[base + 48];
        f[0] = q0.x; f[1] = q0.y; f[2] = q1.x; f[3] = q1.y;
        f[4] = q2.x; f[5] = q2.y; f[6] = q3.x; f[7] = q3.y;
        return f;
    };

    // ---- per-iteration compute: 2 k-substeps x 8 n-tiles = 16 WMMAs,
    //      B fragments software-pipelined one step ahead (rotating regs) ----
    auto compute = [&](const int8_t* tA, const int8_t* tB) {
        v8i afr[2];
#pragma unroll
        for (int ks = 0; ks < 2; ++ks)
            afr[ks] = load_frag(tA, (wave * 16 + frow) * BK + ks * 64 + khalf);

        v8i bcur = load_frag(tB, frow * BK + khalf);  // nt=0, ks=0
#pragma unroll
        for (int ks = 0; ks < 2; ++ks) {
#pragma unroll
            for (int nt = 0; nt < 8; ++nt) {
                v8i bnext = bcur;
                const int nxt = ks * 8 + nt + 1;
                if (nxt < 16) {
                    const int nks = nxt >> 3, nnt = nxt & 7;
                    bnext = load_frag(tB, (nnt * 16 + frow) * BK + nks * 64 + khalf);
                }
                acc[nt] = __builtin_amdgcn_wmma_i32_16x16x64_iu8(
                    true, afr[ks], true, bcur, acc[nt], false, false);
                bcur = bnext;
            }
        }
    };

    const int iters = K / BK;

#if USE_ASYNC_COPY
    // ---- double-buffered LDS, hardware async copies (ASYNCcnt) ----
    auto issue_tile = [&](int k0, int buf) {
#pragma unroll
        for (int q = 0; q < 4; ++q) {
            __builtin_amdgcn_global_load_async_to_lds_b128(
                (v4i_as1*)(gA + k0 + q * 16),
                (v4i_as3*)&lA[buf][ldsoff + q * 16], 0, 0);
            __builtin_amdgcn_global_load_async_to_lds_b128(
                (v4i_as1*)(gB + k0 + q * 16),
                (v4i_as3*)&lB[buf][ldsoff + q * 16], 0, 0);
        }
    };

    issue_tile(0, 0);
    for (int it = 0; it < iters; ++it) {
        const int cur = it & 1;
        __builtin_amdgcn_s_wait_asynccnt(0);  // my copies into buf[cur] done
        __syncthreads();                      // everyone's copies done; prev reads of buf[cur^1] done
        if (it + 1 < iters) issue_tile((it + 1) * BK, cur ^ 1);
        compute(&lA[cur][0], &lB[cur][0]);    // overlaps with in-flight async copies
    }
#else
    // ---- register-staged software pipeline: global latency hides behind WMMAs ----
    v4i rA[4], rB[4];
    auto load_regs = [&](int k0) {
#pragma unroll
        for (int q = 0; q < 4; ++q) {
            rA[q] = *(const v4i*)(gA + k0 + q * 16);
            rB[q] = *(const v4i*)(gB + k0 + q * 16);
        }
    };

    load_regs(0);
    for (int it = 0; it < iters; ++it) {
        __syncthreads();                      // readers of previous tile done
#pragma unroll
        for (int q = 0; q < 4; ++q) {         // waits for loads here, then deposits
            *(v4i*)&lA1[ldsoff + q * 16] = rA[q];
            *(v4i*)&lB1[ldsoff + q * 16] = rB[q];
        }
        __syncthreads();
        if (it + 1 < iters) load_regs((it + 1) * BK);  // issue next tile, no wait
        compute(lA1, lB1);
    }
#endif

    // C/D layout: VGPR j, lanes 0-15 -> M=j,N=lane; lanes 16-31 -> M=j+8,N=lane-16
    const int nlane = lane & 15;
    const int mofs  = (lane >> 4) * 8;
#pragma unroll
    for (int nt = 0; nt < 8; ++nt) {
#pragma unroll
        for (int j = 0; j < 8; ++j) {
            float v = requant((float)acc[nt][j] * scale);
            const int m = bm + wave * 16 + mofs + j;
            const int n = bn + nt * 16 + nlane;
            C[(size_t)m * N + n] = (int8_t)(int)v;
        }
    }
}

// Row softmax over N=8192: one block per row, 32 elements/thread in registers.
// If Yb != nullptr, writes bf16 (q * bf_scale); else writes requantized int8 (in-place safe).
__global__ __launch_bounds__(256) void softmax_i8(
    const int8_t* __restrict__ X, int8_t* __restrict__ Yq,
    __hip_bfloat16* __restrict__ Yb, int N,
    float in_scale, float inv_out_scale, float bf_scale)
{
    const int tid  = threadIdx.x;
    const int wave = tid >> 5;
    const int lane = tid & 31;
    const size_t rowbase = (size_t)blockIdx.x * N;
    const int col0 = tid * 32;

    float vals[32];
    {
        v4i d0 = *(const v4i*)&X[rowbase + col0];
        v4i d1 = *(const v4i*)&X[rowbase + col0 + 16];
#pragma unroll
        for (int i = 0; i < 4; ++i) {
            const int w0 = d0[i], w1 = d1[i];
#pragma unroll
            for (int b = 0; b < 4; ++b) {
                vals[i * 4 + b]      = (float)(int8_t)(w0 >> (8 * b)) * in_scale;
                vals[16 + i * 4 + b] = (float)(int8_t)(w1 >> (8 * b)) * in_scale;
            }
        }
    }

    float m = vals[0];
#pragma unroll
    for (int i = 1; i < 32; ++i) m = fmaxf(m, vals[i]);
#pragma unroll
    for (int off = 16; off > 0; off >>= 1)
        m = fmaxf(m, __shfl_xor(m, off, 32));

    __shared__ float redmax[8];
    __shared__ float redsum[8];
    if (lane == 0) redmax[wave] = m;
    __syncthreads();
    float rm = redmax[0];
#pragma unroll
    for (int i = 1; i < 8; ++i) rm = fmaxf(rm, redmax[i]);

    float s = 0.0f;
#pragma unroll
    for (int i = 0; i < 32; ++i) { vals[i] = __expf(vals[i] - rm); s += vals[i]; }
#pragma unroll
    for (int off = 16; off > 0; off >>= 1)
        s += __shfl_xor(s, off, 32);
    if (lane == 0) redsum[wave] = s;
    __syncthreads();
    float rs = 0.0f;
#pragma unroll
    for (int i = 0; i < 8; ++i) rs += redsum[i];
    const float k = inv_out_scale / rs;

    if (Yb) {
#pragma unroll
        for (int i = 0; i < 32; ++i) {
            const float q = requant(vals[i] * k);
            Yb[rowbase + col0 + i] = __float2bfloat16(q * bf_scale);
        }
    } else {
        v4i o0, o1;
#pragma unroll
        for (int i = 0; i < 4; ++i) {
            int w0 = 0, w1 = 0;
#pragma unroll
            for (int b = 0; b < 4; ++b) {
                w0 |= ((int)requant(vals[i * 4 + b] * k) & 0xff) << (8 * b);
                w1 |= ((int)requant(vals[16 + i * 4 + b] * k) & 0xff) << (8 * b);
            }
            o0[i] = w0; o1[i] = w1;
        }
        *(v4i*)&Yq[rowbase + col0]      = o0;
        *(v4i*)&Yq[rowbase + col0 + 16] = o1;
    }
}

extern "C" void kernel_launch(void* const* d_in, const int* in_sizes, int n_in,
                              void* d_out, int out_size, void* d_ws, size_t ws_size,
                              hipStream_t stream)
{
    const int8_t* x_q  = (const int8_t*)d_in[0];
    const float*  sx   = (const float*)d_in[1];   // scale_x (1 elem, on device)
    const int8_t* w1_q = (const int8_t*)d_in[2];
    const int8_t* w2_q = (const int8_t*)d_in[3];
    __hip_bfloat16* out = (__hip_bfloat16*)d_out;

    int8_t* h = (int8_t*)d_ws;                        // [BATCH, HID_DIM] int8
    int8_t* y = h + (size_t)BATCH * HID_DIM;          // [BATCH, OUT_DIM] int8

    dim3 blk(256);

    // linear1: scale = scale_x * W_SCALE / OUT_SCALE
    dim3 g1(HID_DIM / 128, BATCH / 128);
    gemm_i8_wmma<<<g1, blk, 0, stream>>>(x_q, w1_q, h,
                                         BATCH, HID_DIM, IN_DIM,
                                         sx, W_SCALE / OUT_SCALE);

    // softmax1 (in-place int8)
    softmax_i8<<<dim3(BATCH), blk, 0, stream>>>(h, h, nullptr, HID_DIM,
                                                OUT_SCALE, 1.0f / OUT_SCALE, 0.0f);

    // linear2: scale = OUT_SCALE * W_SCALE / OUT_SCALE (constant)
    dim3 g2(OUT_DIM / 128, BATCH / 128);
    gemm_i8_wmma<<<g2, blk, 0, stream>>>(h, w2_q, y,
                                         BATCH, OUT_DIM, HID_DIM,
                                         nullptr, OUT_SCALE * W_SCALE / OUT_SCALE);

    // softmax2 + final dequant to bf16
    softmax_i8<<<dim3(BATCH), blk, 0, stream>>>(y, nullptr, out, OUT_DIM,
                                                OUT_SCALE, 1.0f / OUT_SCALE, OUT_SCALE);
}
